// MLP_23338852287179
// MI455X (gfx1250) — matrix operand order repaired and checked by reading, not compiled, not run on hardware
//
#include <hip/hip_runtime.h>
#include <stdint.h>

typedef __attribute__((ext_vector_type(16))) __bf16 v16bf;
typedef __attribute__((ext_vector_type(8)))  float  v8f;
typedef int v4i_vec __attribute__((vector_size(16)));
typedef __attribute__((address_space(1))) v4i_vec* gptr_v4i;
typedef __attribute__((address_space(3))) v4i_vec* lptr_v4i;

__device__ __forceinline__ unsigned short f32_to_bf16(float f) {
    uint32_t u = __float_as_uint(f);
    u += 0x7FFFu + ((u >> 16) & 1u);   // round-to-nearest-even
    return (unsigned short)(u >> 16);
}

// Async global->LDS 16B copy (ASYNCcnt-tracked, no VGPR round trip).
__device__ __forceinline__ void async_ld_b128(void* lds_dst, const void* gsrc) {
#if __has_builtin(__builtin_amdgcn_global_load_async_to_lds_b128)
    __builtin_amdgcn_global_load_async_to_lds_b128(
        (gptr_v4i)(void*)(uintptr_t)gsrc,
        (lptr_v4i)lds_dst, 0, 0);
#else
    const unsigned off =
        (unsigned)(uintptr_t)(__attribute__((address_space(3))) void*)lds_dst;
    asm volatile("global_load_async_to_lds_b128 %0, %1, off"
                 :: "v"(off), "v"(gsrc) : "memory");
#endif
}

__device__ __forceinline__ void wait_async_zero() {
    asm volatile("s_wait_asynccnt 0x0" ::: "memory");
}

// ---------------------------------------------------------------------------
// Block-64 symmetric int5 MSE quantize-dequantize -> bf16 weights.
// One wave32 per block: each lane owns 2 consecutive floats (coalesced 256B).
// ---------------------------------------------------------------------------
__global__ __launch_bounds__(256) void quant_mse_bf16(
    const float* __restrict__ w, unsigned short* __restrict__ wq, int nblocks)
{
    const int lane = threadIdx.x & 31;
    const int blk  = blockIdx.x * (blockDim.x >> 5) + (threadIdx.x >> 5);
    if (blk >= nblocks) return;

    const float2 v = ((const float2*)(w + (size_t)blk * 64))[lane];

    float a = fmaxf(fabsf(v.x), fabsf(v.y));
    #pragma unroll
    for (int m = 16; m >= 1; m >>= 1) a = fmaxf(a, __shfl_xor(a, m, 32));

    const float base = fmaxf(a, 1e-8f) * (1.0f / 15.0f);   // QMAX = 15
    float best_err = __builtin_inff();
    float best_scale = base;
    #pragma unroll
    for (int s = 0; s < 16; ++s) {                          // linspace(0.4, 1.0, 16)
        const float scale = base * (0.4f + 0.04f * (float)s);
        const float q0 = fminf(fmaxf(rintf(v.x / scale), -15.f), 15.f);
        const float q1 = fminf(fmaxf(rintf(v.y / scale), -15.f), 15.f);
        const float d0 = q0 * scale - v.x;
        const float d1 = q1 * scale - v.y;
        float e = d0 * d0 + d1 * d1;
        #pragma unroll
        for (int m = 16; m >= 1; m >>= 1) e += __shfl_xor(e, m, 32);
        if (e < best_err) { best_err = e; best_scale = scale; }  // strict <: first wins ties
    }
    const float q0 = fminf(fmaxf(rintf(v.x / best_scale), -15.f), 15.f) * best_scale;
    const float q1 = fminf(fmaxf(rintf(v.y / best_scale), -15.f), 15.f) * best_scale;
    wq[(size_t)blk * 64 + lane * 2    ] = f32_to_bf16(q0);
    wq[(size_t)blk * 64 + lane * 2 + 1] = f32_to_bf16(q1);
}

// ---------------------------------------------------------------------------
// bf16 WMMA GEMM: C[M,N] = A[M,K] * B[N,K]^T   (B row-major N x K, K innermost)
// Workgroup tile 128x128, 8 waves (4 in M x 2 in N), wave tile 32x64
// = 2x4 accumulators of v_wmma_f32_16x16x32_bf16. LDS-staged K=32 slices.
// B tiles (and A tiles when already bf16) move via GLOBAL_LOAD_ASYNC_TO_LDS.
// EPI==0: C := bf16(relu(C)^2)    EPI==1: C := fp32(C)
// ---------------------------------------------------------------------------
template <int EPI, bool A_F32>
__global__ __launch_bounds__(256) void wmma_gemm(
    const void* __restrict__ Aq, const unsigned short* __restrict__ Bw,
    void* __restrict__ C, int M, int N, int K)
{
    constexpr int LDA = 40;                         // 32 + 8 pad, keeps 16B align
    __shared__ __align__(16) unsigned short As[128 * LDA];
    __shared__ __align__(16) unsigned short Bs[128 * LDA];

    const int t    = threadIdx.x;
    const int lane = t & 31;
    const int w    = t >> 5;
    const int wm   = w & 3;                         // 0..3  (M direction)
    const int wn   = w >> 2;                        // 0..1  (N direction)
    const int lr   = lane & 15;
    const int kh   = (lane >> 4) << 3;              // A frag: K half offset (0/8)
    const int kb   = (lane >> 4) << 4;              // unused K half offset (0/16); B frag uses kh

    const int mBase = blockIdx.y * 128;
    const int nBase = blockIdx.x * 128;

    const int lrow = t >> 2;                        // 0..63, two passes cover 128 rows
    const int lcol = (t & 3) << 3;                  // 0,8,16,24

    v8f acc[2][4];
    #pragma unroll
    for (int i = 0; i < 2; ++i)
        #pragma unroll
        for (int j = 0; j < 4; ++j)
            acc[i][j] = (v8f){0.f,0.f,0.f,0.f,0.f,0.f,0.f,0.f};

    union V16 { v16bf v; uint4 u[2]; };

    for (int k0 = 0; k0 < K; k0 += 32) {
        // -------- stage 128x32 A and B tiles into LDS --------
        #pragma unroll
        for (int half = 0; half < 2; ++half) {
            const int row = lrow + half * 64;
            if constexpr (A_F32) {
                // f32 -> bf16 conversion must pass through VGPRs
                const float* src = (const float*)Aq + (size_t)(mBase + row) * K + k0 + lcol;
                const float4 f0 = *(const float4*)src;
                const float4 f1 = *(const float4*)(src + 4);
                union { unsigned short h[8]; uint4 u; } tmp;
                tmp.h[0] = f32_to_bf16(f0.x); tmp.h[1] = f32_to_bf16(f0.y);
                tmp.h[2] = f32_to_bf16(f0.z); tmp.h[3] = f32_to_bf16(f0.w);
                tmp.h[4] = f32_to_bf16(f1.x); tmp.h[5] = f32_to_bf16(f1.y);
                tmp.h[6] = f32_to_bf16(f1.z); tmp.h[7] = f32_to_bf16(f1.w);
                *(uint4*)&As[row * LDA + lcol] = tmp.u;
                __builtin_prefetch(src + 32, 0, 3);   // non-faulting, unconditional
            } else {
                // already bf16: async DMA straight into LDS
                const unsigned short* src =
                    (const unsigned short*)Aq + (size_t)(mBase + row) * K + k0 + lcol;
                async_ld_b128(&As[row * LDA + lcol], src);
            }
            const unsigned short* bsrc = Bw + (size_t)(nBase + row) * K + k0 + lcol;
            async_ld_b128(&Bs[row * LDA + lcol], bsrc);
        }
        wait_async_zero();                           // our async writes landed in LDS
        __syncthreads();                             // publish across waves

        // -------- fragment loads (ISA 16-bit A 16x32 / B 32x16 layouts) --------
        V16 afrag[2], bfrag[4];
        #pragma unroll
        for (int i = 0; i < 2; ++i) {
            const int r = (wm * 32 + i * 16 + lr) * LDA;
            afrag[i].u[0] = *(const uint4*)&As[r + kh];        // K = kh..kh+7
            afrag[i].u[1] = *(const uint4*)&As[r + kh + 16];   // K = kh+16..kh+23
        }
        #pragma unroll
        for (int j = 0; j < 4; ++j) {
            const int r = (wn * 64 + j * 16 + lr) * LDA;
            bfrag[j].u[0] = *(const uint4*)&Bs[r + kh];        // K = kh..kh+7
            bfrag[j].u[1] = *(const uint4*)&Bs[r + kh + 16];   // K = kh+16..kh+23
        }

        #pragma unroll
        for (int i = 0; i < 2; ++i)
            #pragma unroll
            for (int j = 0; j < 4; ++j)
                acc[i][j] = __builtin_amdgcn_wmma_f32_16x16x32_bf16(
                    false, afrag[i].v, false, bfrag[j].v,
                    (short)0, acc[i][j], false, false);
        __syncthreads();
    }

    // -------- epilogue: C/D layout -> VGPR r: M = r + (lane<16 ? 0 : 8), N = lane%16
    const int mRow = mBase + wm * 32 + ((lane >> 4) << 3);
    const int nCol = nBase + wn * 64 + lr;
    #pragma unroll
    for (int i = 0; i < 2; ++i) {
        #pragma unroll
        for (int j = 0; j < 4; ++j) {
            #pragma unroll
            for (int r = 0; r < 8; ++r) {
                float vv = acc[i][j][r];
                const size_t idx = (size_t)(mRow + i * 16 + r) * N + (nCol + j * 16);
                if constexpr (EPI == 0) {
                    vv = fmaxf(vv, 0.0f);            // relu
                    vv *= vv;                        // square
                    ((unsigned short*)C)[idx] = f32_to_bf16(vv);
                } else {
                    ((float*)C)[idx] = vv;
                }
            }
        }
    }
}

// ---------------------------------------------------------------------------
// x: (4,2048,2048) f32   w_fc: (8192,2048) f32   w_proj: (2048,8192) f32
// out: (4,2048,2048) f32
// Workspace: wfc_bf16 32MB | wproj_bf16 32MB | h2_bf16 128MB  (192MB total)
// ---------------------------------------------------------------------------
extern "C" void kernel_launch(void* const* d_in, const int* in_sizes, int n_in,
                              void* d_out, int out_size, void* d_ws, size_t ws_size,
                              hipStream_t stream)
{
    const float* x      = (const float*)d_in[0];
    const float* w_fc   = (const float*)d_in[1];
    const float* w_proj = (const float*)d_in[2];
    float* out = (float*)d_out;

    const int M = 8192;   // 4 * 2048 tokens
    const int D = 2048;   // model dim
    const int H = 8192;   // hidden dim

    char* ws = (char*)d_ws;
    unsigned short* wfc_bf   = (unsigned short*)ws;                         // H*D
    unsigned short* wproj_bf = (unsigned short*)(ws + (size_t)H * D * 2);   // D*H
    unsigned short* h2       = (unsigned short*)(ws + (size_t)H * D * 4);   // M*H

    const int nblocks = H * D / 64;                 // 262144 blocks for each weight
    quant_mse_bf16<<<nblocks / 8, 256, 0, stream>>>(w_fc,   wfc_bf,   nblocks);
    quant_mse_bf16<<<nblocks / 8, 256, 0, stream>>>(w_proj, wproj_bf, nblocks);

    // h2 = (relu(x @ wq_fc^T))^2  stored bf16
    wmma_gemm<0, true ><<<dim3(H / 128, M / 128), 256, 0, stream>>>(x,  wfc_bf,   h2,  M, H, D);
    // out = h2 @ wq_proj^T  stored fp32
    wmma_gemm<1, false><<<dim3(D / 128, M / 128), 256, 0, stream>>>(h2, wproj_bf, out, M, D, H);
}
